// mul_pointnet2_AT_pred_55121610277155
// MI455X (gfx1250) — compile-verified, matrix operand order checked by reading, not run on hardware
//
#include <hip/hip_runtime.h>
#include <stdint.h>

typedef unsigned short u16;
typedef __attribute__((ext_vector_type(8)))  __bf16 v8bf;
typedef __attribute__((ext_vector_type(16))) __bf16 v16bf;
typedef __attribute__((ext_vector_type(8)))  float  v8f;

#define MT      32      // rows per block in the fused per-point kernel
#define NWAVES  8       // 256 threads / wave32

// ---------- bf16 helpers (hardware convert) ----------
__device__ __forceinline__ u16 f2bf(float f) {
  __bf16 h = (__bf16)f;                    // v_cvt_*_bf16_f32 (RNE)
  return __builtin_bit_cast(u16, h);
}
__device__ __forceinline__ float bf2f(u16 h) {
  unsigned int u = ((unsigned int)h) << 16;
  return __builtin_bit_cast(float, u);
}
__device__ __forceinline__ float bf2f_lo(unsigned int u) {
  return __builtin_bit_cast(float, u << 16);
}
__device__ __forceinline__ float bf2f_hi(unsigned int u) {
  return __builtin_bit_cast(float, u & 0xFFFF0000u);
}
__device__ __forceinline__ v8bf ld8(const u16* p) {
  return __builtin_bit_cast(v8bf, *reinterpret_cast<const uint4*>(p));
}

// ---------- generic fused GEMM layer: D = relu(A @ Wt^T * s + t) ----------
// Compile-time K/N/M so the k-loop fully unrolls and loads clause-batch.
template<int KPAD, int NOUT, int MTILE>
__device__ __forceinline__ void mlp_gemm(
    const u16* sA, const u16* gW,
    const float* sc, const float* bi,
    u16* sD, int Wd, int colOfs)
{
  const int lane = threadIdx.x & 31;
  const int wave = threadIdx.x >> 5;
  const int rsub = lane & 15;
  const int hi8  = (lane >> 4) << 3;           // 0 for lanes 0-15, 8 for 16-31
  constexpr int RT = MTILE >> 4;
  constexpr int NTILES = RT * (NOUT >> 4);
  for (int tile = wave; tile < NTILES; tile += NWAVES) {
    const int r0 = (tile % RT) << 4;
    const int c0 = (tile / RT) << 4;
    v8f acc = {};
    const u16* aRow = sA + (r0 + rsub) * KPAD + hi8;
    const u16* bRow = gW + (c0 + rsub) * KPAD + hi8;
#pragma unroll
    for (int k0 = 0; k0 < KPAD; k0 += 32) {
      v8bf alo = ld8(aRow + k0);
      v8bf ahi = ld8(aRow + k0 + 16);
      v8bf blo = ld8(bRow + k0);
      v8bf bhi = ld8(bRow + k0 + 16);
      v16bf a = __builtin_shufflevector(alo, ahi, 0,1,2,3,4,5,6,7,8,9,10,11,12,13,14,15);
      v16bf b = __builtin_shufflevector(blo, bhi, 0,1,2,3,4,5,6,7,8,9,10,11,12,13,14,15);
      acc = __builtin_amdgcn_wmma_f32_16x16x32_bf16(false, a, false, b, (short)0, acc, false, false);
    }
    const int n = c0 + rsub;
    const float s = sc[n], t = bi[n];
    const int rb = r0 + hi8;
#pragma unroll
    for (int j = 0; j < 8; ++j) {
      float y = acc[j] * s + t;
      y = y > 0.f ? y : 0.f;
      sD[(rb + j) * Wd + colOfs + n] = f2bf(y);
    }
  }
}

// ---------- prep kernels ----------
__global__ __launch_bounds__(256) void kprep_w(const float* __restrict__ W, u16* __restrict__ Wt,
                                               int K, int N, int Kpad) {
  int idx = blockIdx.x * 256 + threadIdx.x;
  if (idx >= N * Kpad) return;
  int n = idx / Kpad, k = idx % Kpad;
  float v = (k < K) ? W[k * N + n] : 0.f;
  Wt[idx] = f2bf(v);
}
__global__ __launch_bounds__(256) void kprep_bn(const float* g, const float* b, const float* m,
                                                const float* v, float* s, float* t, int n) {
  int i = blockIdx.x * 256 + threadIdx.x;
  if (i >= n) return;
  float sc = g[i] * rsqrtf(v[i] + 1e-5f);
  s[i] = sc; t[i] = b[i] - m[i] * sc;
}

// ---------- fused per-point chain ----------
struct MainArgs {
  const float *xyz, *points;
  const u16 *Wt1, *Wt2, *Wt3, *Wt4, *Wt5, *Wt6;
  const float *s1,*t1,*s2,*t2,*s3,*t3,*s4,*t4,*s5,*t5,*s6,*t6;
  const float *att1w, *att1b, *att3w;
  u16* gG;          // [65536][1024] bf16
  float* logits;    // [65536]
};

__global__ __launch_bounds__(256) void kmain(MainArgs A) {
  __shared__ __align__(16) u16 buf0[MT * 256];   // 16 KB
  __shared__ __align__(16) u16 buf1[MT * 512];   // 32 KB
  __shared__ float sXYZ[MT * 3];
  __shared__ float sGate[MT];
  __shared__ float sLogit[MT];
  const int tid = threadIdx.x;
  const int R = blockIdx.x * MT;

  // stage input: [xyz | points | 0-pad] into 32-wide buf0
  for (int idx = tid; idx < MT * 32; idx += 256) {
    int r = idx >> 5, c = idx & 31;
    float v = 0.f;
    if (c < 3)      { v = A.xyz[(size_t)(R + r) * 3 + c]; sXYZ[r * 3 + c] = v; }
    else if (c < 6) { v = A.points[(size_t)(R + r) * 3 + (c - 3)]; }
    buf0[idx] = f2bf(v);
  }
  __syncthreads();
  mlp_gemm< 32,  64, MT>(buf0, A.Wt1, A.s1, A.t1, buf1,  64, 0);  __syncthreads();
  mlp_gemm< 64, 128, MT>(buf1, A.Wt2, A.s2, A.t2, buf0, 128, 0);  __syncthreads();
  mlp_gemm<128, 256, MT>(buf0, A.Wt3, A.s3, A.t3, buf1, 288, 3);  __syncthreads();

  // per-point sigmoid gate over ungated f (cols 3..258 of buf1)
  if (tid < MT) {
    float acc = A.att1b[0];
    for (int k = 0; k < 256; ++k) acc += bf2f(buf1[tid * 288 + 3 + k]) * A.att1w[k];
    sGate[tid] = 1.f / (1.f + expf(-acc));
  }
  // write xyz into cols 0..2 and zero pad cols 259..287
  for (int idx = tid; idx < MT * 32; idx += 256) {
    int r = idx >> 5, c = idx & 31;
    if (c < 3) buf1[r * 288 + c] = f2bf(sXYZ[r * 3 + c]);
    else if (c - 3 < 29) buf1[r * 288 + 259 + (c - 3)] = 0;
  }
  __syncthreads();
  for (int idx = tid; idx < MT * 256; idx += 256) {
    int r = idx >> 8, k = idx & 255;
    buf1[r * 288 + 3 + k] = f2bf(bf2f(buf1[r * 288 + 3 + k]) * sGate[r]);
  }
  __syncthreads();
  mlp_gemm<288, 256, MT>(buf1, A.Wt4, A.s4, A.t4, buf0, 256, 0);  __syncthreads();
  mlp_gemm<256, 512, MT>(buf0, A.Wt5, A.s5, A.t5, buf1, 512, 0);  __syncthreads();
  if (tid < MT) sLogit[tid] = 0.f;
  __syncthreads();

  // final layer 512 -> 1024: stream bf16 g to HBM + accumulate attention logits.
  // Per-lane logit partials live in registers (16 rows per lane), flushed once.
  {
    const int lane = tid & 31;
    const int wave = tid >> 5;
    const int rsub = lane & 15;
    const int hi8  = (lane >> 4) << 3;
    float part[16];
#pragma unroll
    for (int p = 0; p < 16; ++p) part[p] = 0.f;
    for (int tile = wave; tile < 2 * 64; tile += NWAVES) {
      const int r0 = (tile & 1) << 4;
      const int c0 = (tile >> 1) << 4;
      v8f acc = {};
      const u16* aRow = buf1 + (r0 + rsub) * 512 + hi8;
      const u16* bRow = A.Wt6 + (c0 + rsub) * 512 + hi8;
#pragma unroll
      for (int k0 = 0; k0 < 512; k0 += 32) {
        v8bf alo = ld8(aRow + k0);
        v8bf ahi = ld8(aRow + k0 + 16);
        v8bf blo = ld8(bRow + k0);
        v8bf bhi = ld8(bRow + k0 + 16);
        v16bf a = __builtin_shufflevector(alo, ahi, 0,1,2,3,4,5,6,7,8,9,10,11,12,13,14,15);
        v16bf b = __builtin_shufflevector(blo, bhi, 0,1,2,3,4,5,6,7,8,9,10,11,12,13,14,15);
        acc = __builtin_amdgcn_wmma_f32_16x16x32_bf16(false, a, false, b, (short)0, acc, false, false);
      }
      const int n = c0 + rsub;
      const float s = A.s6[n], t = A.t6[n], aw = A.att3w[n];
      const int rb = r0 + hi8;
      const int pb = (tile & 1) ? 8 : 0;
#pragma unroll
      for (int j = 0; j < 8; ++j) {
        float y = acc[j] * s + t;
        y = y > 0.f ? y : 0.f;
        A.gG[(size_t)(R + rb + j) * 1024 + n] = f2bf(y);
        part[pb + j] += y * aw;
      }
    }
#pragma unroll
    for (int p = 0; p < 16; ++p) {
      const int row = (p < 8 ? p : p + 8) + hi8;   // p<8: rows 0..7(+hi8); p>=8: rows 16..23(+hi8)
      atomicAdd(&sLogit[row], part[p]);
    }
  }
  __syncthreads();
  if (tid < MT) A.logits[R + tid] = sLogit[tid];
}

// ---------- per-batch softmax + attention-weighted pooling ----------
__global__ __launch_bounds__(256) void ksoftpool(const float* __restrict__ logits,
                                                 const u16* __restrict__ gG,
                                                 float* __restrict__ pooled) {
  __shared__ float sExp[4096];
  __shared__ float sRed[256];
  __shared__ float sMax, sSum;
  const int b = blockIdx.x, tid = threadIdx.x;
  const float* lg = logits + b * 4096;
  float m = -3.4e38f;
  for (int r = tid; r < 4096; r += 256) m = fmaxf(m, lg[r]);
  sRed[tid] = m; __syncthreads();
  if (tid == 0) { float mm = sRed[0]; for (int i = 1; i < 256; ++i) mm = fmaxf(mm, sRed[i]); sMax = mm; }
  __syncthreads();
  const float mx = sMax;
  float sum = 0.f;
  for (int r = tid; r < 4096; r += 256) { float e = expf(lg[r] - mx); sExp[r] = e; sum += e; }
  sRed[tid] = sum; __syncthreads();
  if (tid == 0) { float ss = 0.f; for (int i = 0; i < 256; ++i) ss += sRed[i]; sSum = ss; }
  __syncthreads();
  const float inv = 1.f / sSum;
  // thread owns packed column pairs: cols {2t,2t+1} and {512+2t, 513+2t}
  float a0 = 0.f, a1 = 0.f, a2 = 0.f, a3 = 0.f;
  const unsigned int* gb = reinterpret_cast<const unsigned int*>(gG + (size_t)b * 4096 * 1024);
  for (int r = 0; r < 4096; ++r) {
    const float w = sExp[r];
    const unsigned int* row = gb + (size_t)r * 512;
    unsigned int u0 = row[tid];
    unsigned int u1 = row[tid + 256];
    a0 += w * bf2f_lo(u0);
    a1 += w * bf2f_hi(u0);
    a2 += w * bf2f_lo(u1);
    a3 += w * bf2f_hi(u1);
  }
  pooled[b * 1024 + 2 * tid]       = a0 * inv;
  pooled[b * 1024 + 2 * tid + 1]   = a1 * inv;
  pooled[b * 1024 + 512 + 2 * tid]     = a2 * inv;
  pooled[b * 1024 + 512 + 2 * tid + 1] = a3 * inv;
}

// ---------- head: [16,1024] -> 512 -> 256 -> 1 ----------
struct HeadArgs {
  const float* pooled;
  const u16 *WtF1, *WtF2;
  const float *sh1, *th1, *sh2, *th2;
  const float *predw, *predb;
  float* out;
};
__global__ __launch_bounds__(256) void khead(HeadArgs A) {
  __shared__ __align__(16) u16 sP[16 * 1024];   // 32 KB
  __shared__ __align__(16) u16 sH1[16 * 512];   // 16 KB
  __shared__ __align__(16) u16 sH2[16 * 256];   // 8 KB
  const int tid = threadIdx.x;
  for (int idx = tid; idx < 16 * 1024; idx += 256) sP[idx] = f2bf(A.pooled[idx]);
  __syncthreads();
  mlp_gemm<1024, 512, 16>(sP,  A.WtF1, A.sh1, A.th1, sH1, 512, 0);  __syncthreads();
  mlp_gemm< 512, 256, 16>(sH1, A.WtF2, A.sh2, A.th2, sH2, 256, 0);  __syncthreads();
  if (tid < 16) {
    float acc = A.predb[0];
    for (int k = 0; k < 256; ++k) acc += bf2f(sH2[tid * 256 + k]) * A.predw[k];
    A.out[tid] = acc;
  }
}

// ---------- host ----------
extern "C" void kernel_launch(void* const* d_in, const int* in_sizes, int n_in,
                              void* d_out, int out_size, void* d_ws, size_t ws_size,
                              hipStream_t stream) {
  const float* xyz     = (const float*)d_in[0];
  const float* points  = (const float*)d_in[1];
  const float* sa1_w1  = (const float*)d_in[2];
  const float* sa1_w2  = (const float*)d_in[7];
  const float* sa1_w3  = (const float*)d_in[12];
  const float* sa1_aw  = (const float*)d_in[17];
  const float* sa1_ab  = (const float*)d_in[18];
  const float* sa3_w1  = (const float*)d_in[19];
  const float* sa3_w2  = (const float*)d_in[24];
  const float* sa3_w3  = (const float*)d_in[29];
  const float* sa3_aw  = (const float*)d_in[34];
  const float* fc1_w   = (const float*)d_in[35];
  const float* fc2_w   = (const float*)d_in[40];
  const float* pred_w  = (const float*)d_in[45];
  const float* pred_b  = (const float*)d_in[46];
  (void)in_sizes; (void)n_in; (void)out_size; (void)ws_size;

  size_t off = 0;
  auto alloc = [&](size_t bytes) -> void* {
    void* p = (char*)d_ws + off;
    off += (bytes + 255) & ~(size_t)255;
    return p;
  };
  u16* Wt1  = (u16*)alloc(  64u *   32 * 2);
  u16* Wt2  = (u16*)alloc( 128u *   64 * 2);
  u16* Wt3  = (u16*)alloc( 256u *  128 * 2);
  u16* Wt4  = (u16*)alloc( 256u *  288 * 2);
  u16* Wt5  = (u16*)alloc( 512u *  256 * 2);
  u16* Wt6  = (u16*)alloc(1024u *  512 * 2);
  u16* WtF1 = (u16*)alloc( 512u * 1024 * 2);
  u16* WtF2 = (u16*)alloc( 256u *  512 * 2);
  float* s1 = (float*)alloc(  64 * 4); float* t1 = (float*)alloc(  64 * 4);
  float* s2 = (float*)alloc( 128 * 4); float* t2 = (float*)alloc( 128 * 4);
  float* s3 = (float*)alloc( 256 * 4); float* t3 = (float*)alloc( 256 * 4);
  float* s4 = (float*)alloc( 256 * 4); float* t4 = (float*)alloc( 256 * 4);
  float* s5 = (float*)alloc( 512 * 4); float* t5 = (float*)alloc( 512 * 4);
  float* s6 = (float*)alloc(1024 * 4); float* t6 = (float*)alloc(1024 * 4);
  float* sh1= (float*)alloc( 512 * 4); float* th1= (float*)alloc( 512 * 4);
  float* sh2= (float*)alloc( 256 * 4); float* th2= (float*)alloc( 256 * 4);
  float* logits = (float*)alloc(65536u * 4);
  float* pooled = (float*)alloc(16u * 1024 * 4);
  u16*   gG     = (u16*)alloc((size_t)65536 * 1024 * 2);

  auto prepW = [&](const float* W, u16* Wt, int K, int N, int Kpad) {
    int total = N * Kpad;
    kprep_w<<<(total + 255) / 256, 256, 0, stream>>>(W, Wt, K, N, Kpad);
  };
  auto prepBN = [&](int base, float* s, float* t, int n) {
    kprep_bn<<<(n + 255) / 256, 256, 0, stream>>>(
        (const float*)d_in[base], (const float*)d_in[base + 1],
        (const float*)d_in[base + 2], (const float*)d_in[base + 3], s, t, n);
  };
  prepW(sa1_w1,  Wt1,    6,   64,   32);
  prepW(sa1_w2,  Wt2,   64,  128,   64);
  prepW(sa1_w3,  Wt3,  128,  256,  128);
  prepW(sa3_w1,  Wt4,  259,  256,  288);
  prepW(sa3_w2,  Wt5,  256,  512,  256);
  prepW(sa3_w3,  Wt6,  512, 1024,  512);
  prepW(fc1_w,   WtF1,1024,  512, 1024);
  prepW(fc2_w,   WtF2, 512,  256,  512);
  prepBN( 3, s1, t1,   64);
  prepBN( 8, s2, t2,  128);
  prepBN(13, s3, t3,  256);
  prepBN(20, s4, t4,  256);
  prepBN(25, s5, t5,  512);
  prepBN(30, s6, t6, 1024);
  prepBN(36, sh1, th1, 512);
  prepBN(41, sh2, th2, 256);

  MainArgs MA;
  MA.xyz = xyz; MA.points = points;
  MA.Wt1 = Wt1; MA.Wt2 = Wt2; MA.Wt3 = Wt3; MA.Wt4 = Wt4; MA.Wt5 = Wt5; MA.Wt6 = Wt6;
  MA.s1 = s1; MA.t1 = t1; MA.s2 = s2; MA.t2 = t2; MA.s3 = s3; MA.t3 = t3;
  MA.s4 = s4; MA.t4 = t4; MA.s5 = s5; MA.t5 = t5; MA.s6 = s6; MA.t6 = t6;
  MA.att1w = sa1_aw; MA.att1b = sa1_ab; MA.att3w = sa3_aw;
  MA.gG = gG; MA.logits = logits;
  kmain<<<65536 / MT, 256, 0, stream>>>(MA);

  ksoftpool<<<16, 256, 0, stream>>>(logits, gG, pooled);

  HeadArgs HA;
  HA.pooled = pooled; HA.WtF1 = WtF1; HA.WtF2 = WtF2;
  HA.sh1 = sh1; HA.th1 = th1; HA.sh2 = sh2; HA.th2 = th2;
  HA.predw = pred_w; HA.predb = pred_b;
  HA.out = (float*)d_out;
  khead<<<1, 256, 0, stream>>>(HA);
}
